// GraphTransformer_26809185862281
// MI455X (gfx1250) — compile-verified, matrix operand order checked by reading, not run on hardware
//
#include <hip/hip_runtime.h>

typedef __attribute__((ext_vector_type(16))) _Float16 v16h;
typedef __attribute__((ext_vector_type(8)))  float    v8f;
typedef unsigned int uint32;

#define NN   50000
#define EE   800000
#define FIN  128
#define HIDD 128
#define NH   4
#define DD   32
#define NG   4
#define MT   5          // M-tiles (of 16 rows) per wave: 80 rows; 50000 = 625*80

// ---------------------------------------------------------------------------
// helpers
// ---------------------------------------------------------------------------
__device__ __forceinline__ void atomicMaxFloat(float* addr, float val) {
    // mixed-sign float max via int/uint atomics (init must be -inf)
    if (val >= 0.0f) atomicMax((int*)addr, __float_as_int(val));
    else             atomicMin((unsigned int*)addr, __float_as_uint(val));
}

// ---------------------------------------------------------------------------
// conversions
// ---------------------------------------------------------------------------
__global__ void cvt_f32_to_f16_kernel(const float* __restrict__ src,
                                      _Float16* __restrict__ dst, int n) {
    int id = blockIdx.x * blockDim.x + threadIdx.x;
    if (id < n) dst[id] = (_Float16)src[id];
}

// W [128(in) x 128(out)] row-major f32  ->  WT [128(out) x 128(in)] f16
__global__ void cvt_w_transpose_kernel(const float* __restrict__ W,
                                       _Float16* __restrict__ WT) {
    int id = blockIdx.x * blockDim.x + threadIdx.x;   // 0..16383
    int i = id >> 7;          // input dim
    int o = id & 127;         // output dim
    WT[o * 128 + i] = (_Float16)W[i * 128 + o];
}

// ---------------------------------------------------------------------------
// fused 4-projection WMMA GEMM:  out_z = X16 @ W_z + b_z   (z = q,k,v,skip)
// one wave per (5 x 16)-row x 16-col block. All 4 B (weight) fragments are
// preloaded once; each M-tile then runs a 4-deep WMMA accumulate chain whose
// A fragments load as one clause, with the next tile's loads pipelining
// underneath the current chain. 20 WMMAs / wave, weight tile fetched once.
// ---------------------------------------------------------------------------
__global__ __launch_bounds__(32)
void gemm4_wmma_kernel(const _Float16* __restrict__ X16,          // [N,128]
                       const _Float16* __restrict__ WT,           // 4 x [128x128] (transposed)
                       const float* __restrict__ b0, const float* __restrict__ b1,
                       const float* __restrict__ b2, const float* __restrict__ b3,
                       float* __restrict__ o0, float* __restrict__ o1,
                       float* __restrict__ o2, float* __restrict__ o3) {
    const int z = blockIdx.z;
    const _Float16* W   = WT + (size_t)z * 128 * 128;
    const float* bias   = (z == 0) ? b0 : (z == 1) ? b1 : (z == 2) ? b2 : b3;
    float*       out    = (z == 0) ? o0 : (z == 1) ? o1 : (z == 2) ? o2 : o3;

    const int l  = threadIdx.x;            // lane 0..31
    const int n  = l & 15;                 // tile column (A: tile row)
    const int kb = (l < 16) ? 0 : 8;       // per-lane K sub-base (16-bit A/B layout)
    const int rowbase = blockIdx.x * (16 * MT);
    const int colbase = blockIdx.y * 16;

    // ---- preload the full 16-col weight tile: 4 K-step fragments ----------
    const uint32* brow = (const uint32*)(W + (size_t)(colbase + n) * 128);     // f16 pairs
    union { v16h h; uint32 u[8]; } Bf[4];
#pragma unroll
    for (int kt = 0; kt < 4; ++kt) {
#pragma unroll
        for (int v = 0; v < 8; ++v) {
            const int k = kt * 32 + kb + 2 * (v & 3) + ((v >= 4) ? 16 : 0);
            Bf[kt].u[v] = brow[k >> 1];
        }
    }

    const int m0 = (l < 16) ? 0 : 8;
    const float bv = bias[colbase + n];

#pragma unroll
    for (int mt = 0; mt < MT; ++mt) {
        const uint32* arow = (const uint32*)(X16 + (size_t)(rowbase + mt * 16 + n) * 128);
        // all 4 A fragments for this M-tile: independent, issue as one clause
        union { v16h h; uint32 u[8]; } Af[4];
#pragma unroll
        for (int kt = 0; kt < 4; ++kt) {
#pragma unroll
            for (int v = 0; v < 8; ++v) {
                const int k = kt * 32 + kb + 2 * (v & 3) + ((v >= 4) ? 16 : 0);
                Af[kt].u[v] = arow[k >> 1];
            }
        }
        v8f c = {};
#pragma unroll
        for (int kt = 0; kt < 4; ++kt)
            c = __builtin_amdgcn_wmma_f32_16x16x32_f16(false, Af[kt].h, false, Bf[kt].h,
                                                       (short)0, c, false, false);
        // store this tile immediately -> only one live accumulator
#pragma unroll
        for (int i = 0; i < 8; ++i) {
            const int r = rowbase + mt * 16 + m0 + i;
            out[(size_t)r * 128 + colbase + n] = c[i] + bv;
        }
    }
}

// ---------------------------------------------------------------------------
// edge phase
// ---------------------------------------------------------------------------
__global__ void init_ms_kernel(float* __restrict__ M, float* __restrict__ S, int n) {
    int id = blockIdx.x * blockDim.x + threadIdx.x;
    if (id < n) { M[id] = -__builtin_huge_valf(); S[id] = 0.0f; }
}

// pass 1: logits[e,h] = <q[dst,h,:], k[src,h,:]> / sqrt(D);  segment max over dst
__global__ void edge_logits_kernel(const float* __restrict__ Q,
                                   const float* __restrict__ K,
                                   const float* __restrict__ V,
                                   const long long* __restrict__ ei,
                                   float* __restrict__ LG,
                                   float* __restrict__ M) {
    int id = blockIdx.x * blockDim.x + threadIdx.x;   // over E*H
    if (id >= EE * NH) return;
    const int e = id >> 2;
    const int h = id & 3;
    const int src = (int)ei[e];
    const int dst = (int)ei[EE + e];
    const float4* qa = (const float4*)(Q + (size_t)dst * HIDD + h * DD);
    const float4* ka = (const float4*)(K + (size_t)src * HIDD + h * DD);
    // warm L2 for the alpha*v scatter pass
    __builtin_prefetch((const void*)(V + (size_t)src * HIDD + h * DD), 0, 1);
    float acc = 0.0f;
#pragma unroll
    for (int j = 0; j < 8; ++j) {
        float4 q4 = qa[j], k4 = ka[j];
        acc += q4.x * k4.x + q4.y * k4.y + q4.z * k4.z + q4.w * k4.w;
    }
    const float lg = acc * 0.17677669529663689f;   // 1/sqrt(32)
    LG[id] = lg;
    atomicMaxFloat(&M[dst * NH + h], lg);
}

// pass 2: e = exp(logit - m[dst]); segment sum over dst
__global__ void edge_expsum_kernel(const long long* __restrict__ ei,
                                   float* __restrict__ LG,
                                   const float* __restrict__ M,
                                   float* __restrict__ S) {
    int id = blockIdx.x * blockDim.x + threadIdx.x;   // over E*H
    if (id >= EE * NH) return;
    const int e = id >> 2;
    const int h = id & 3;
    const int dst = (int)ei[EE + e];
    const float v = __expf(LG[id] - M[dst * NH + h]);
    LG[id] = v;
    atomicAdd(&S[dst * NH + h], v);
}

// pass 3: agg[dst,c] += (e/s[dst,h]) * v[src,c]   (agg pre-loaded with skip)
__global__ void edge_aggregate_kernel(const long long* __restrict__ ei,
                                      const float* __restrict__ LG,
                                      const float* __restrict__ S,
                                      const float* __restrict__ V,
                                      float* __restrict__ AGG) {
    long long id = (long long)blockIdx.x * blockDim.x + threadIdx.x;   // over E*128
    if (id >= (long long)EE * HIDD) return;
    const int e = (int)(id >> 7);
    const int c = (int)(id & 127);
    const int h = c >> 5;
    const int src = (int)ei[e];
    const int dst = (int)ei[EE + e];
    const float alpha = LG[e * NH + h] / S[dst * NH + h];
    atomicAdd(&AGG[(size_t)dst * HIDD + c], alpha * V[(size_t)src * HIDD + c]);
}

// h = relu(agg) in place, plus f16 copy for the next layer's WMMA A operand
__global__ void relu_cvt_kernel(float* __restrict__ AGG,
                                _Float16* __restrict__ H16, int n) {
    int id = blockIdx.x * blockDim.x + threadIdx.x;
    if (id < n) {
        float v = fmaxf(AGG[id], 0.0f);
        AGG[id] = v;
        H16[id] = (_Float16)v;
    }
}

// ---------------------------------------------------------------------------
// pooling + FC head
// ---------------------------------------------------------------------------
__global__ void init_pool_kernel(float* __restrict__ PL, float* __restrict__ CNT) {
    int id = threadIdx.x;
    for (int i = id; i < NG * HIDD; i += 256) PL[i] = 0.0f;
    if (id < NG) CNT[id] = 0.0f;
}

__global__ void pool_sum_kernel(const float* __restrict__ Hf,
                                const long long* __restrict__ batch,
                                float* __restrict__ PL) {
    long long id = (long long)blockIdx.x * blockDim.x + threadIdx.x;   // over N*128
    if (id >= (long long)NN * HIDD) return;
    const int nidx = (int)(id >> 7);
    const int c = (int)(id & 127);
    const int g = (int)batch[nidx];
    atomicAdd(&PL[g * HIDD + c], Hf[id]);
}

__global__ void pool_cnt_kernel(const long long* __restrict__ batch,
                                float* __restrict__ CNT) {
    int id = blockIdx.x * blockDim.x + threadIdx.x;
    if (id < NN) atomicAdd(&CNT[(int)batch[id]], 1.0f);
}

__global__ void fc_out_kernel(const float* __restrict__ PL,
                              const float* __restrict__ CNT,
                              const float* __restrict__ Wfc,
                              const float* __restrict__ bfc,
                              float* __restrict__ out) {
    __shared__ float red[NG][HIDD];
    const int c = threadIdx.x;                 // 0..127
    const float w = Wfc[c];
#pragma unroll
    for (int g = 0; g < NG; ++g) {
        const float cnt = fmaxf(CNT[g], 1.0f);
        red[g][c] = (PL[g * HIDD + c] / cnt) * w;
    }
    __syncthreads();
    if (c < NG) {
        float a = 0.0f;
        for (int j = 0; j < HIDD; ++j) a += red[c][j];
        out[c] = a + bfc[0];
    }
}

// ---------------------------------------------------------------------------
// launcher
// ---------------------------------------------------------------------------
extern "C" void kernel_launch(void* const* d_in, const int* in_sizes, int n_in,
                              void* d_out, int out_size, void* d_ws, size_t ws_size,
                              hipStream_t stream) {
    const float*     x     = (const float*)d_in[0];
    const long long* ei    = (const long long*)d_in[1];
    const long long* batch = (const long long*)d_in[2];
    const float* Wm[8] = { (const float*)d_in[3],  (const float*)d_in[5],
                           (const float*)d_in[7],  (const float*)d_in[9],
                           (const float*)d_in[11], (const float*)d_in[13],
                           (const float*)d_in[15], (const float*)d_in[17] };
    const float* bm[8] = { (const float*)d_in[4],  (const float*)d_in[6],
                           (const float*)d_in[8],  (const float*)d_in[10],
                           (const float*)d_in[12], (const float*)d_in[14],
                           (const float*)d_in[16], (const float*)d_in[18] };
    const float* Wfc = (const float*)d_in[19];
    const float* bfc = (const float*)d_in[20];
    float* out = (float*)d_out;

    // workspace carve-up (256B aligned slices)
    char*  base = (char*)d_ws;
    size_t off  = 0;
    auto carve = [&](size_t bytes) -> void* {
        void* p = base + off;
        off += (bytes + 255) & ~(size_t)255;
        return p;
    };
    _Float16* X16 = (_Float16*)carve((size_t)NN * HIDD * sizeof(_Float16));   // activations f16
    _Float16* WT  = (_Float16*)carve((size_t)8 * 128 * 128 * sizeof(_Float16));
    float* Q   = (float*)carve((size_t)NN * HIDD * sizeof(float));
    float* Kf  = (float*)carve((size_t)NN * HIDD * sizeof(float));
    float* Vf  = (float*)carve((size_t)NN * HIDD * sizeof(float));
    float* AGG = (float*)carve((size_t)NN * HIDD * sizeof(float));
    float* LG  = (float*)carve((size_t)EE * NH * sizeof(float));
    float* M   = (float*)carve((size_t)NN * NH * sizeof(float));
    float* S   = (float*)carve((size_t)NN * NH * sizeof(float));
    float* PL  = (float*)carve((size_t)NG * HIDD * sizeof(float));
    float* CNT = (float*)carve((size_t)NG * sizeof(float));
    (void)ws_size; (void)n_in; (void)in_sizes; (void)out_size;

    const int T = 256;
    const int nfeat = NN * HIDD;

    // --- prep: f16 activations + transposed f16 weights ---------------------
    cvt_f32_to_f16_kernel<<<(nfeat + T - 1) / T, T, 0, stream>>>(x, X16, nfeat);
    for (int w = 0; w < 8; ++w)
        cvt_w_transpose_kernel<<<(128 * 128) / T, T, 0, stream>>>(Wm[w], WT + (size_t)w * 128 * 128);

    const dim3 ggrid(NN / (16 * MT), HIDD / 16, 4);   // 625 x 8 x 4, one wave each
    const int  eh_blocks  = (EE * NH + T - 1) / T;
    const long long ec    = (long long)EE * HIDD;
    const int  ec_blocks  = (int)((ec + T - 1) / T);

    for (int layer = 0; layer < 2; ++layer) {
        const _Float16* WTl = WT + (size_t)layer * 4 * 128 * 128;
        const float** b = &bm[layer * 4];
        // Q,K,V + skip(->AGG) projections on the matrix pipe
        gemm4_wmma_kernel<<<ggrid, 32, 0, stream>>>(X16, WTl,
                                                    b[0], b[1], b[2], b[3],
                                                    Q, Kf, Vf, AGG);
        init_ms_kernel<<<(NN * NH + T - 1) / T, T, 0, stream>>>(M, S, NN * NH);
        edge_logits_kernel<<<eh_blocks, T, 0, stream>>>(Q, Kf, Vf, ei, LG, M);
        edge_expsum_kernel<<<eh_blocks, T, 0, stream>>>(ei, LG, M, S);
        edge_aggregate_kernel<<<ec_blocks, T, 0, stream>>>(ei, LG, S, Vf, AGG);
        relu_cvt_kernel<<<(nfeat + T - 1) / T, T, 0, stream>>>(AGG, X16, nfeat);
    }

    // --- mean pool per graph + FC head --------------------------------------
    init_pool_kernel<<<1, 256, 0, stream>>>(PL, CNT);
    pool_sum_kernel<<<(nfeat + T - 1) / T, T, 0, stream>>>(AGG, batch, PL);
    pool_cnt_kernel<<<(NN + T - 1) / T, T, 0, stream>>>(batch, CNT);
    fc_out_kernel<<<1, HIDD, 0, stream>>>(PL, CNT, Wfc, bfc, out);
}
